// LocalConcatSheafLearner_81484119540399
// MI455X (gfx1250) — matrix operand order repaired and checked by reading, not compiled, not run on hardware
//
#include <hip/hip_runtime.h>

// CDNA5 WMMA vector types
typedef __attribute__((ext_vector_type(16))) __bf16 v16bf;
typedef __attribute__((ext_vector_type(8)))  float  v8f;

#define IN_CH   64
#define NOUT    25
#define TILE_E  16   // edges per wave tile (WMMA M)

// gfx1250 has a hardware V_TANH_F32 transcendental; use it when the
// toolchain exposes the builtin, else fall back to libm tanhf.
__device__ __forceinline__ float fast_tanh(float v) {
#if __has_builtin(__builtin_amdgcn_tanhf)
    return __builtin_amdgcn_tanhf(v);
#else
    return tanhf(v);
#endif
}

__global__ __launch_bounds__(256) void sheaf_wmma_kernel(
    const float* __restrict__ x,        // [N_NODES, 64]
    const long long* __restrict__ ei,   // [2, E] int64
    const float* __restrict__ W,        // [128, 25] row-major
    float* __restrict__ out,            // [E, 25]
    int nEdges, int nTiles)
{
    const int lane  = threadIdx.x & 31;
    const int wSlot = threadIdx.x >> 5;
    const int wid   = blockIdx.x * (blockDim.x >> 5) + wSlot;
    const int nWav  = gridDim.x * (blockDim.x >> 5);
    const int m     = lane & 15;   // row-in-tile (A) / column-in-tile (B,C,D)
    const int hi    = lane >> 4;   // lane half selects K sub-range

    // ---- Load B (= W, bf16) fragments once per wave; reused for all tiles.
    // B as packed here (contiguous K, same order as A): lanes 0-15 hold K=0..15 of column N=lane,
    // lanes 16-31 hold K=16..31; two K-values packed per VGPR in ascending K.
    v16bf B[4][2];
#pragma unroll
    for (int ks = 0; ks < 4; ++ks) {
#pragma unroll
        for (int nt = 0; nt < 2; ++nt) {
            const int ng = nt * 16 + m;           // global output column
#pragma unroll
            for (int i = 0; i < 16; ++i) {
                const int k = ks * 32 + hi * 16 + i;   // global K (0..127)
                const float w = (ng < NOUT) ? W[k * NOUT + ng] : 0.0f;
                B[ks][nt][i] = (__bf16)w;
            }
        }
    }

    for (int tile = wid; tile < nTiles; tile += nWav) {
        const int eBase = tile * TILE_E;

        // Prefetch next tile's edge indices (streamed; emits global_prefetch_b8)
        {
            const int nb = (tile + nWav) < nTiles ? (tile + nWav) * TILE_E : eBase;
            __builtin_prefetch(&ei[nb + m], 0, 1);
            __builtin_prefetch(&ei[(size_t)nEdges + nb + m], 0, 1);
        }

        // Each half-wave lane pair (lane, lane+16) serves edge eBase+m.
        int e = eBase + m;
        const int ec = (e < nEdges) ? e : (nEdges - 1);   // clamp for tail
        const long long r = __builtin_nontemporal_load(&ei[ec]);
        const long long c = __builtin_nontemporal_load(&ei[(size_t)nEdges + ec]);
        const float* __restrict__ xr = x + (size_t)r * IN_CH;
        const float* __restrict__ xc = x + (size_t)c * IN_CH;

        v8f acc0 = {};  // output columns 0..15
        v8f acc1 = {};  // output columns 16..24 (padded to 31)

#pragma unroll
        for (int ks = 0; ks < 4; ++ks) {
            // K-slices 0,1 read x[row]; slices 2,3 read x[col]
            const float* src  = (ks < 2) ? xr : xc;
            const int    foff = (ks & 1) * 32;
            // A layout (16x32, 16-bit): lane half 'hi' covers local K
            // {hi*16..hi*16+15} contiguously (matches B).
            const float* s0 = src + foff + hi * 16;
            const float* s1 = src + foff + hi * 16 + 8;
            const float4 p0 = *(const float4*)(s0);
            const float4 p1 = *(const float4*)(s0 + 4);
            const float4 q0 = *(const float4*)(s1);
            const float4 q1 = *(const float4*)(s1 + 4);

            v16bf a;
            a[0]  = (__bf16)p0.x; a[1]  = (__bf16)p0.y;
            a[2]  = (__bf16)p0.z; a[3]  = (__bf16)p0.w;
            a[4]  = (__bf16)p1.x; a[5]  = (__bf16)p1.y;
            a[6]  = (__bf16)p1.z; a[7]  = (__bf16)p1.w;
            a[8]  = (__bf16)q0.x; a[9]  = (__bf16)q0.y;
            a[10] = (__bf16)q0.z; a[11] = (__bf16)q0.w;
            a[12] = (__bf16)q1.x; a[13] = (__bf16)q1.y;
            a[14] = (__bf16)q1.z; a[15] = (__bf16)q1.w;

            acc0 = __builtin_amdgcn_wmma_f32_16x16x32_bf16(
                       false, a, false, B[ks][0], (short)0, acc0, false, false);
            acc1 = __builtin_amdgcn_wmma_f32_16x16x32_bf16(
                       false, a, false, B[ks][1], (short)0, acc1, false, false);
        }

        // C/D layout (16x16 f32): lane L, VGPR j -> row M = j + 8*hi, col N = L&15.
        // tanh (HW trans op) + nontemporal store (output is 320MB, keep x in L2).
#pragma unroll
        for (int j = 0; j < 8; ++j) {
            const int eo = eBase + j + 8 * hi;
            if (eo < nEdges) {
                float* o = out + (size_t)eo * NOUT;
                __builtin_nontemporal_store(fast_tanh(acc0[j]), o + m);
                if (m + 16 < NOUT) {
                    __builtin_nontemporal_store(fast_tanh(acc1[j]), o + m + 16);
                }
            }
        }
    }
}

extern "C" void kernel_launch(void* const* d_in, const int* in_sizes, int n_in,
                              void* d_out, int out_size, void* d_ws, size_t ws_size,
                              hipStream_t stream) {
    const float*     x  = (const float*)d_in[0];
    const long long* ei = (const long long*)d_in[1];   // int64 per reference
    const float*     W  = (const float*)d_in[2];
    float*           o  = (float*)d_out;

    const int nEdges = in_sizes[1] / 2;
    const int nTiles = (nEdges + TILE_E - 1) / TILE_E;

    // 256 threads = 8 waves/block; grid-stride over edge tiles.
    int blocks = 2048;
    const int wavesNeeded = nTiles;
    if (blocks * 8 > wavesNeeded) blocks = (wavesNeeded + 7) / 8;
    if (blocks < 1) blocks = 1;

    hipLaunchKernelGGL(sheaf_wmma_kernel, dim3(blocks), dim3(256), 0, stream,
                       x, ei, W, o, nEdges, nTiles);
}
